// GRUCell_80032420593692
// MI455X (gfx1250) — matrix operand order repaired and checked by reading, not compiled, not run on hardware
//
#include <hip/hip_runtime.h>

typedef __attribute__((ext_vector_type(16))) _Float16 v16h;
typedef __attribute__((ext_vector_type(8)))  _Float16 v8h;
typedef __attribute__((ext_vector_type(8)))  float    v8f;
typedef __attribute__((ext_vector_type(4)))  float    v4f;
typedef int v4i_vs __attribute__((vector_size(16)));   // matches builtin param type

#define BB   4096   // batch
#define NIN  1024   // input dim (== hidden dim here)
#define NH   1024   // hidden dim
#define BM   128    // block tile M
#define BN   64     // block tile N
#define BK   32     // K step (f16 WMMA depth)
#define LDT  40     // padded f16 row stride in LDS (80B: 16B-aligned, bank-spread)
#define NSTEP (NIN / BK)                            // 32
#define STAGE_ELEMS (2 * BM * LDT + 6 * BN * LDT)   // 25600 f16 = 50KB per stage

// workspace layout (f16 elements)
#define WSB_ELEMS ((size_t)6 * 16 * 32 * 64 * 32)   // [m][nt][kt][n64][k32] = 12MB
#define WSX_ELEMS ((size_t)BB * NIN)                // 8MB
#define WSH_ELEMS ((size_t)BB * NH)                 // 8MB
#define WS_NEEDED ((WSB_ELEMS + WSX_ELEMS + WSH_ELEMS) * 2)

// ---- CDNA5 async global->LDS path (guarded: falls back to manual copies) ----
#if defined(__gfx1250__) && __has_builtin(__builtin_amdgcn_global_load_async_to_lds_b128)
#define USE_ASYNC_LDS 1
#else
#define USE_ASYNC_LDS 0
#endif

#if USE_ASYNC_LDS
#if __has_builtin(__builtin_amdgcn_s_wait_asynccnt)
#define WAIT_ASYNC(n) __builtin_amdgcn_s_wait_asynccnt(n)
#else
#define WAIT_ASYNC(n) asm volatile("s_wait_asynccnt %0" :: "i"(n) : "memory")
#endif
// builtin signature: (v4i AS1* src, v4i AS3* dst, imm ioffset, imm cpol);
// IOFFSET (bytes) is added to BOTH the global and LDS addresses.
#define ASYNC_CP16(g, l, off)                                                  \
  __builtin_amdgcn_global_load_async_to_lds_b128(                              \
      (__attribute__((address_space(1))) v4i_vs*)(g),                          \
      (__attribute__((address_space(3))) v4i_vs*)(l), (off), 0)
#endif

__device__ __forceinline__ v16h cat16(v8h lo, v8h hi) {
  return __builtin_shufflevector(lo, hi, 0,1,2,3,4,5,6,7,8,9,10,11,12,13,14,15);
}

__device__ __forceinline__ v8f wmma_f16(v16h a, v16h b, v8f c) {
  return __builtin_amdgcn_wmma_f32_16x16x32_f16(false, a, false, b, (short)0, c,
                                                false, false);
}

// cut(x) = clamp(floor(x / 256), -128, 127)  (kept in float, as in reference)
__device__ __forceinline__ float cutf(float v) {
  return fminf(fmaxf(floorf(v * (1.0f / 256.0f)), -128.0f), 127.0f);
}

__device__ __forceinline__ float sigm(float t) {
  return 1.0f / (1.0f + __expf(-t));
}

// ---------------------------------------------------------------------------
// Pre-pass 1: weights f32 -> f16, transposed + tiled.
// Output tile (m, nt, kt) is 64(n) x 32(k), k contiguous, 4KB contiguous chunk.
// ---------------------------------------------------------------------------
extern "C" __global__ __launch_bounds__(256, 1)
void convert_w_kernel(const float* __restrict__ w_in, const float* __restrict__ w_h,
                      _Float16* __restrict__ wB)
{
  __shared__ __align__(128) _Float16 tile[64 * LDT];   // [n][k], padded

  const int bid = blockIdx.x;          // 0 .. 6*16*32-1
  const int m   = bid >> 9;
  const int rem = bid & 511;
  const int nt  = rem >> 5;
  const int kt  = rem & 31;
  const int tid = threadIdx.x;

  const float* src = (m < 3) ? (w_in + (size_t)m * NIN * NH)
                             : (w_h  + (size_t)(m - 3) * NH * NH);

  {
    const int k  = tid >> 3;
    const int ng = tid & 7;
    const float* p = src + (size_t)(kt * 32 + k) * NH + nt * 64 + ng * 8;
    #pragma unroll
    for (int i = 0; i < 8; ++i)
      tile[(ng * 8 + i) * LDT + k] = (_Float16)p[i];
  }
  __syncthreads();
  {
    const int n  = tid >> 2;
    const int kg = tid & 3;
    v8h v = *(const v8h*)&tile[n * LDT + kg * 8];
    _Float16* dst = wB + (((size_t)(m * 16 + nt) * 32 + kt) * 2048);
    *(v8h*)(dst + (size_t)tid * 8) = v;   // n*32 + kg*8 == tid*8
  }
}

// ---------------------------------------------------------------------------
// Pre-pass 2: x and hid f32 -> f16, layout preserved (K contiguous).
// ---------------------------------------------------------------------------
extern "C" __global__ __launch_bounds__(256, 1)
void convert_xh_kernel(const float* __restrict__ x, const float* __restrict__ hid,
                       _Float16* __restrict__ wX, _Float16* __restrict__ wH)
{
  const size_t i = ((size_t)blockIdx.x * 256 + threadIdx.x) * 8;
  v4f a0 = *(const v4f*)(x + i),   a1 = *(const v4f*)(x + i + 4);
  v4f b0 = *(const v4f*)(hid + i), b1 = *(const v4f*)(hid + i + 4);
  v8h xo, ho;
  #pragma unroll
  for (int j = 0; j < 4; ++j) {
    xo[j] = (_Float16)a0[j]; xo[4 + j] = (_Float16)a1[j];
    ho[j] = (_Float16)b0[j]; ho[4 + j] = (_Float16)b1[j];
  }
  *(v8h*)(wX + i) = xo;
  *(v8h*)(wH + i) = ho;
}

// ---------------------------------------------------------------------------
// Main fused GRU kernel, f16 operands from workspace (fast path).
// Block tile 128x64, 8 waves (4m x 2n), each wave 32x32 via 2x2 WMMA subtiles.
// Async path: double-buffered LDS stage filled by GLOBAL_LOAD_ASYNC_TO_LDS,
// next stage prefetched under the current stage's WMMAs (ASYNCcnt pipeline).
// ---------------------------------------------------------------------------
extern "C" __global__ __launch_bounds__(256, 1)
void gru_wmma_f16_kernel(const _Float16* __restrict__ wX, const _Float16* __restrict__ wH,
                         const _Float16* __restrict__ wB,
                         const float* __restrict__ hid,
                         const float* __restrict__ b_in, const float* __restrict__ b_h,
                         float* __restrict__ out)
{
#if USE_ASYNC_LDS
  __shared__ __align__(128) _Float16 smem[2 * STAGE_ELEMS];   // 100KB (320KB WGP LDS)
#else
  __shared__ __align__(128) _Float16 smem[STAGE_ELEMS];
#endif

  const int tid  = threadIdx.x;
  const int wave = tid >> 5;
  const int lane = tid & 31;
  const int half = lane >> 4;
  const int l16  = lane & 15;
  const int wm   = wave >> 1;   // 0..3
  const int wn   = wave & 1;    // 0..1

  const int bm  = blockIdx.y * BM;
  const int bn  = blockIdx.x * BN;
  const int bnt = blockIdx.x;

  v8f acc[4][2][2];
  #pragma unroll
  for (int g = 0; g < 4; ++g)
    #pragma unroll
    for (int mi = 0; mi < 2; ++mi)
      #pragma unroll
      for (int ni = 0; ni < 2; ++ni)
        acc[g][mi][ni] = (v8f){0.f,0.f,0.f,0.f,0.f,0.f,0.f,0.f};

  const int arow_ld = tid >> 1;        // A-tile row 0..127
  const int ahalf   = tid & 1;         // 16-elem K chunk
  const int bdst_n  = tid >> 2;        // B-tile dest col 0..63
  const int bdst_k  = (tid & 3) * 8;

  // fragment compute for one K-step from a given stage buffer
  auto compute_tile = [&](const _Float16* sAx, const _Float16* sAh, const _Float16* sB) {
    v16h ax[2], ah[2];
    #pragma unroll
    for (int mi = 0; mi < 2; ++mi) {
      const _Float16* pa = sAx + (wm * 32 + mi * 16 + l16) * LDT + half * 8;
      ax[mi] = cat16(*(const v8h*)pa, *(const v8h*)(pa + 16));
      const _Float16* ph = sAh + (wm * 32 + mi * 16 + l16) * LDT + half * 8;
      ah[mi] = cat16(*(const v8h*)ph, *(const v8h*)(ph + 16));
    }
    #pragma unroll
    for (int ni = 0; ni < 2; ++ni) {
      const int ncol = wn * 32 + ni * 16 + l16;
      #pragma unroll
      for (int m = 0; m < 6; ++m) {
        const _Float16* pb = sB + ((size_t)m * BN + ncol) * LDT + half * 8;
        v16h bf = cat16(*(const v8h*)pb, *(const v8h*)(pb + 16));
        const int g = (m == 0 || m == 3) ? 0 : (m == 1 || m == 4) ? 1 : (m == 2) ? 2 : 3;
        #pragma unroll
        for (int mi = 0; mi < 2; ++mi)
          acc[g][mi][ni] = wmma_f16((m < 3) ? ax[mi] : ah[mi], bf, acc[g][mi][ni]);
      }
    }
  };

#if USE_ASYNC_LDS
  // issue exactly 10 async b128 copies per thread per stage
  auto stage = [&](int buf, int k0v) {
    _Float16* bAx = smem + buf * STAGE_ELEMS;
    _Float16* bAh = bAx + BM * LDT;
    _Float16* bB  = bAx + 2 * BM * LDT;
    const _Float16* gx = wX + (size_t)(bm + arow_ld) * NIN + k0v + ahalf * 16;
    const _Float16* gh = wH + (size_t)(bm + arow_ld) * NH  + k0v + ahalf * 16;
    _Float16* dx = bAx + arow_ld * LDT + ahalf * 16;
    _Float16* dh = bAh + arow_ld * LDT + ahalf * 16;
    ASYNC_CP16(gx, dx, 0);  ASYNC_CP16(gx, dx, 16);   // IOFFSET applies to both sides
    ASYNC_CP16(gh, dh, 0);  ASYNC_CP16(gh, dh, 16);
    const int kt = k0v >> 5;
    #pragma unroll
    for (int m = 0; m < 6; ++m) {
      const _Float16* tsrc = wB + (((size_t)(m * 16 + bnt) * 32 + kt) * 2048) + (size_t)tid * 8;
      _Float16* dstB = bB + ((size_t)m * BN + bdst_n) * LDT + bdst_k;
      ASYNC_CP16(tsrc, dstB, 0);
    }
  };

  stage(0, 0);
  for (int it = 0; it < NSTEP; ++it) {
    const int cur = it & 1;
    __syncthreads();                       // all waves done reading buf[1-cur] (it-1)
    if (it + 1 < NSTEP) {
      stage(1 - cur, (it + 1) * BK);       // prefetch next stage under this step's WMMAs
      WAIT_ASYNC(10);                      // in-order completion: ours for buf[cur] done
    } else {
      WAIT_ASYNC(0);
    }
    __syncthreads();                       // everyone's buf[cur] writes visible
    const _Float16* sAx = smem + cur * STAGE_ELEMS;
    compute_tile(sAx, sAx + BM * LDT, sAx + 2 * BM * LDT);
  }
#else
  _Float16* sAx = smem;
  _Float16* sAh = smem + BM * LDT;
  _Float16* sB  = smem + 2 * BM * LDT;
  for (int k0 = 0; k0 < NIN; k0 += BK) {
    const int kt = k0 >> 5;
    __syncthreads();
    {
      const v8h* gx = (const v8h*)(wX + (size_t)(bm + arow_ld) * NIN + k0 + ahalf * 16);
      const v8h* gh = (const v8h*)(wH + (size_t)(bm + arow_ld) * NH  + k0 + ahalf * 16);
      v8h x0 = gx[0], x1 = gx[1];
      v8h h0 = gh[0], h1 = gh[1];
      _Float16* dx = sAx + arow_ld * LDT + ahalf * 16;
      _Float16* dh = sAh + arow_ld * LDT + ahalf * 16;
      *(v8h*)dx = x0; *(v8h*)(dx + 8) = x1;
      *(v8h*)dh = h0; *(v8h*)(dh + 8) = h1;
    }
    #pragma unroll
    for (int m = 0; m < 6; ++m) {
      const _Float16* tsrc = wB + (((size_t)(m * 16 + bnt) * 32 + kt) * 2048);
      v8h v = *(const v8h*)(tsrc + (size_t)tid * 8);
      *(v8h*)(sB + ((size_t)m * BN + bdst_n) * LDT + bdst_k) = v;
    }
    __syncthreads();
    compute_tile(sAx, sAh, sB);
  }
#endif

  // ---- epilogue: exact reference gate math on f32 accumulators ----
  #pragma unroll
  for (int ni = 0; ni < 2; ++ni) {
    const int col = bn + wn * 32 + ni * 16 + l16;
    const float bs0 = b_in[col]          + b_h[col];
    const float bs1 = b_in[NH + col]     + b_h[NH + col];
    const float bi2 = b_in[2 * NH + col];
    const float bh2 = b_h[2 * NH + col];
    #pragma unroll
    for (int mi = 0; mi < 2; ++mi) {
      const int rbase = bm + wm * 32 + mi * 16 + half * 8;
      #pragma unroll
      for (int j = 0; j < 8; ++j) {
        const int row = rbase + j;
        const float g0  = acc[0][mi][ni][j] + bs0;
        const float r   = floorf(127.0f * sigm(cutf(g0) * (1.0f / 40.0f)));
        const float g1  = acc[1][mi][ni][j] + bs1;
        const float z   = floorf(127.0f * sigm(cutf(g1) * (1.0f / 40.0f)));
        const float g7  = acc[2][mi][ni][j] + bi2;
        const float g8c = cutf(acc[3][mi][ni][j] + bh2);
        const float tt  = cutf(g7 + r * g8c) * (1.0f / 64.0f);
        const float nn  = floorf(127.0f * (2.0f / (1.0f + __expf(-2.0f * tt)) - 1.0f));
        const float omz = cutf(127.0f - z);
        const float hv  = hid[(size_t)row * NH + col];
        out[(size_t)row * NH + col] = cutf(nn * omz + hv * z);
      }
    }
  }
}

// ---------------------------------------------------------------------------
// Fallback: all-in-one kernel (f32 global operands, inline cvt), used only
// if the workspace is too small for the f16 staging buffers.
// ---------------------------------------------------------------------------
extern "C" __global__ __launch_bounds__(256, 1)
void gru_wmma_fused_kernel(const float* __restrict__ x,    const float* __restrict__ hid,
                           const float* __restrict__ w_in, const float* __restrict__ w_h,
                           const float* __restrict__ b_in, const float* __restrict__ b_h,
                           float* __restrict__ out)
{
  __shared__ __align__(128) _Float16 smem[STAGE_ELEMS];
  _Float16* sAx = smem;
  _Float16* sAh = smem + BM * LDT;
  _Float16* sB  = smem + 2 * BM * LDT;

  const int tid  = threadIdx.x;
  const int wave = tid >> 5;
  const int lane = tid & 31;
  const int half = lane >> 4;
  const int l16  = lane & 15;
  const int wm   = wave >> 1;
  const int wn   = wave & 1;

  const int bm = blockIdx.y * BM;
  const int bn = blockIdx.x * BN;

  v8f acc[4][2][2];
  #pragma unroll
  for (int g = 0; g < 4; ++g)
    #pragma unroll
    for (int mi = 0; mi < 2; ++mi)
      #pragma unroll
      for (int ni = 0; ni < 2; ++ni)
        acc[g][mi][ni] = (v8f){0.f,0.f,0.f,0.f,0.f,0.f,0.f,0.f};

  const int arow_ld = tid >> 1;
  const int ahalf   = tid & 1;
  const int bn_t    = tid & 63;
  const int bg_t    = tid >> 6;

  for (int k0 = 0; k0 < NIN; k0 += BK) {
    __syncthreads();
    {
      const v4f* gx = (const v4f*)(x   + (size_t)(bm + arow_ld) * NIN + k0 + ahalf * 16);
      const v4f* gh = (const v4f*)(hid + (size_t)(bm + arow_ld) * NH  + k0 + ahalf * 16);
      v4f x0 = gx[0], x1 = gx[1], x2 = gx[2], x3 = gx[3];
      v4f h0 = gh[0], h1 = gh[1], h2 = gh[2], h3 = gh[3];
      v8h xl, xh, hl, hh;
      #pragma unroll
      for (int i = 0; i < 4; ++i) {
        xl[i] = (_Float16)x0[i]; xl[4 + i] = (_Float16)x1[i];
        xh[i] = (_Float16)x2[i]; xh[4 + i] = (_Float16)x3[i];
        hl[i] = (_Float16)h0[i]; hl[4 + i] = (_Float16)h1[i];
        hh[i] = (_Float16)h2[i]; hh[4 + i] = (_Float16)h3[i];
      }
      _Float16* dx = sAx + arow_ld * LDT + ahalf * 16;
      _Float16* dh = sAh + arow_ld * LDT + ahalf * 16;
      *(v8h*)dx = xl; *(v8h*)(dx + 8) = xh;
      *(v8h*)dh = hl; *(v8h*)(dh + 8) = hh;
    }
    #pragma unroll
    for (int m = 0; m < 6; ++m) {
      const float* src  = (m < 3) ? (w_in + (size_t)m * NIN * NH)
                                  : (w_h  + (size_t)(m - 3) * NH * NH);
      const float* colp = src + (size_t)(k0 + bg_t * 8) * NH + bn + bn_t;
      v8h v;
      #pragma unroll
      for (int j = 0; j < 8; ++j) v[j] = (_Float16)colp[(size_t)j * NH];
      *(v8h*)(sB + ((size_t)m * BN + bn_t) * LDT + bg_t * 8) = v;
    }
    __syncthreads();

    v16h ax[2], ah[2];
    #pragma unroll
    for (int mi = 0; mi < 2; ++mi) {
      const _Float16* pa = sAx + (wm * 32 + mi * 16 + l16) * LDT + half * 8;
      ax[mi] = cat16(*(const v8h*)pa, *(const v8h*)(pa + 16));
      const _Float16* ph = sAh + (wm * 32 + mi * 16 + l16) * LDT + half * 8;
      ah[mi] = cat16(*(const v8h*)ph, *(const v8h*)(ph + 16));
    }
    #pragma unroll
    for (int ni = 0; ni < 2; ++ni) {
      const int ncol = wn * 32 + ni * 16 + l16;
      #pragma unroll
      for (int m = 0; m < 6; ++m) {
        const _Float16* pb = sB + ((size_t)m * BN + ncol) * LDT + half * 8;
        v16h bf = cat16(*(const v8h*)pb, *(const v8h*)(pb + 16));
        const int g = (m == 0 || m == 3) ? 0 : (m == 1 || m == 4) ? 1 : (m == 2) ? 2 : 3;
        #pragma unroll
        for (int mi = 0; mi < 2; ++mi)
          acc[g][mi][ni] = wmma_f16((m < 3) ? ax[mi] : ah[mi], bf, acc[g][mi][ni]);
      }
    }
  }

  #pragma unroll
  for (int ni = 0; ni < 2; ++ni) {
    const int col = bn + wn * 32 + ni * 16 + l16;
    const float bs0 = b_in[col]          + b_h[col];
    const float bs1 = b_in[NH + col]     + b_h[NH + col];
    const float bi2 = b_in[2 * NH + col];
    const float bh2 = b_h[2 * NH + col];
    #pragma unroll
    for (int mi = 0; mi < 2; ++mi) {
      const int rbase = bm + wm * 32 + mi * 16 + half * 8;
      #pragma unroll
      for (int j = 0; j < 8; ++j) {
        const int row = rbase + j;
        const float g0  = acc[0][mi][ni][j] + bs0;
        const float r   = floorf(127.0f * sigm(cutf(g0) * (1.0f / 40.0f)));
        const float g1  = acc[1][mi][ni][j] + bs1;
        const float z   = floorf(127.0f * sigm(cutf(g1) * (1.0f / 40.0f)));
        const float g7  = acc[2][mi][ni][j] + bi2;
        const float g8c = cutf(acc[3][mi][ni][j] + bh2);
        const float tt  = cutf(g7 + r * g8c) * (1.0f / 64.0f);
        const float nn  = floorf(127.0f * (2.0f / (1.0f + __expf(-2.0f * tt)) - 1.0f));
        const float omz = cutf(127.0f - z);
        const float hv  = hid[(size_t)row * NH + col];
        out[(size_t)row * NH + col] = cutf(nn * omz + hv * z);
      }
    }
  }
}

extern "C" void kernel_launch(void* const* d_in, const int* in_sizes, int n_in,
                              void* d_out, int out_size, void* d_ws, size_t ws_size,
                              hipStream_t stream) {
  (void)in_sizes; (void)n_in; (void)out_size;
  const float* x    = (const float*)d_in[0];
  const float* hid  = (const float*)d_in[1];
  const float* w_in = (const float*)d_in[2];   // (3, 1024, 1024)
  const float* w_h  = (const float*)d_in[3];   // (3, 1024, 1024)
  const float* b_in = (const float*)d_in[4];   // (3, 1024)
  const float* b_h  = (const float*)d_in[5];   // (3, 1024)
  float* out = (float*)d_out;                  // (4096, 1024) f32

  dim3 grid(NH / BN, BB / BM);                 // (16, 32)

  if (ws_size >= WS_NEEDED) {
    _Float16* wB = (_Float16*)d_ws;
    _Float16* wX = wB + WSB_ELEMS;
    _Float16* wH = wX + WSX_ELEMS;
    convert_w_kernel <<<6 * 16 * 32, 256, 0, stream>>>(w_in, w_h, wB);
    convert_xh_kernel<<<(BB * NIN) / (256 * 8), 256, 0, stream>>>(x, hid, wX, wH);
    gru_wmma_f16_kernel<<<grid, 256, 0, stream>>>(wX, wH, wB, hid, b_in, b_h, out);
  } else {
    gru_wmma_fused_kernel<<<grid, 256, 0, stream>>>(x, hid, w_in, w_h, b_in, b_h, out);
  }
}
